// Attention_30700426231909
// MI455X (gfx1250) — matrix operand order repaired and checked by reading, not compiled, not run on hardware
//
#include <hip/hip_runtime.h>
#include <math.h>

#define B_  32
#define S_  4096
#define DV_ 256
#define DQ_ 256
#define H_  256

typedef __bf16 v16bf __attribute__((ext_vector_type(16)));
typedef __bf16 v2bf  __attribute__((ext_vector_type(2)));
typedef float  v8f   __attribute__((ext_vector_type(8)));

#if __has_builtin(__builtin_amdgcn_tanhf)
#define TANHF(x) __builtin_amdgcn_tanhf(x)
#elif __has_builtin(__builtin_amdgcn_tanh_f32)
#define TANHF(x) __builtin_amdgcn_tanh_f32(x)
#else
#define TANHF(x) tanhf(x)
#endif

static __device__ __forceinline__ unsigned short f2bf_rne(float f) {
    unsigned int x = __float_as_uint(f);
    unsigned int r = x + 0x7fffu + ((x >> 16) & 1u);   // round-to-nearest-even
    return (unsigned short)(r >> 16);
}

// Pack two f32 -> two bf16 (uses v_cvt_pk_bf16_f32 when available).
static __device__ __forceinline__ v2bf pack2(float a, float b) {
#if __has_builtin(__builtin_amdgcn_cvt_pk_bf16_f32)
    return __builtin_amdgcn_cvt_pk_bf16_f32(a, b);
#else
    union { unsigned int i; v2bf v; } u;
    u.i = (unsigned int)f2bf_rne(a) | ((unsigned int)f2bf_rne(b) << 16);
    return u.v;
#endif
}

union BF16Frag {
    v16bf v;
    v2bf  pk[8];
    unsigned int w[8];
};

// ---------------------------------------------------------------------------
// Kernel 1a: qpb[b][h] = bias[h] + dot(query[b,:], W2[h,:])
// ---------------------------------------------------------------------------
__global__ void __launch_bounds__(256) qpb_kernel(const float* __restrict__ query,
                                                  const float* __restrict__ W2,
                                                  const float* __restrict__ bias,
                                                  float* __restrict__ qpb) {
    const int b = blockIdx.x;
    const int h = threadIdx.x;
    const float* q  = query + (size_t)b * DQ_;
    const float* w2 = W2 + (size_t)h * DQ_;
    float acc = bias[h];
    #pragma unroll 8
    for (int k = 0; k < DQ_; ++k) acc += q[k] * w2[k];
    qpb[b * H_ + h] = acc;
}

// ---------------------------------------------------------------------------
// Kernel 1b: W1 (f32 [H][DV]) -> bf16 row-major [H][DV]
// ---------------------------------------------------------------------------
__global__ void __launch_bounds__(256) convw1_kernel(const float* __restrict__ W1,
                                                     unsigned short* __restrict__ W1bf) {
    const int idx = blockIdx.x * 256 + threadIdx.x;      // 16384 threads, 4 elems each
    const float4 f = ((const float4*)W1)[idx];
    union { v2bf v; unsigned int i; } lo, hi;
    lo.v = pack2(f.x, f.y);
    hi.v = pack2(f.z, f.w);
    ((unsigned int*)W1bf)[idx * 2 + 0] = lo.i;
    ((unsigned int*)W1bf)[idx * 2 + 1] = hi.i;
}

// ---------------------------------------------------------------------------
// Kernel 2: scores[b][s] = tanh(V[b,s,:]@W1^T + qpb[b,:]) @ w
//
// 16 waves/WG; wave = TWO 16-row s-subtiles (A strips in regs, 128 VGPRs).
// W1 (bf16, 128KB) staged once per WG in LDS; wave loops over 16 h-chunks.
// Each B fragment read from LDS feeds two WMMAs (independent accumulators).
// The B fragment is software-pipelined one step ahead so each WMMA pair
// waits on a load issued a full iteration earlier (no LDS-latency bubble).
// ---------------------------------------------------------------------------
__global__ void __launch_bounds__(512, 1) score_kernel(const float* __restrict__ V,
                                                       const unsigned short* __restrict__ W1bf,
                                                       const float* __restrict__ qpb,
                                                       const float* __restrict__ wvec,
                                                       float* __restrict__ scores) {
    extern __shared__ char smem[];
    unsigned short* w1s = (unsigned short*)smem;                  // 128 KB
    float*          qs  = (float*)(smem + H_ * DV_ * 2);          //   1 KB
    float*          wsd = qs + H_;                                //   1 KB

    const int b    = blockIdx.y;
    const int wave = (int)threadIdx.x >> 5;
    const int lane = (int)threadIdx.x & 31;
    const int lh   = lane & 15;
    const int grp  = lane >> 4;                                   // 0: lanes 0-15

    // ---- stage W1 (bf16) + qpb row + w into LDS ----
    {
        const uint4* src = (const uint4*)W1bf;
        uint4*       dst = (uint4*)w1s;
        for (int i = (int)threadIdx.x; i < (H_ * DV_) / 8; i += 512)
            dst[i] = src[i];
        if (threadIdx.x < H_) {
            qs[threadIdx.x]  = qpb[b * H_ + threadIdx.x];
            wsd[threadIdx.x] = wvec[threadIdx.x];
        }
    }
    __syncthreads();

    // ---- load & convert this wave's two A strips (2 x 16 rows x 256 K) ----
    const int s0 = blockIdx.x * 512 + wave * 32;                  // first strip base row
    BF16Frag afrag0[8], afrag1[8];
    {
        const int row0 = s0 + lh;                                 // strip 0, A row M = lh
        const int row1 = s0 + 16 + lh;                            // strip 1
        const float4* vp0 = (const float4*)(V + ((size_t)b * S_ + row0) * DV_);
        const float4* vp1 = (const float4*)(V + ((size_t)b * S_ + row1) * DV_);
        #pragma unroll
        for (int kc = 0; kc < 8; ++kc) {
            // K order used here (contiguous, B matches): grp0 lanes hold K 0-15, grp1: 16-31
            const int c0 = kc * 8 + grp * 4;                      // float4 idx of K = kc*32+grp*16
            float4 f0 = vp0[c0];
            float4 f1 = vp0[c0 + 1];
            float4 f2 = vp0[c0 + 2];                              // +8 K values
            float4 f3 = vp0[c0 + 3];
            afrag0[kc].pk[0] = pack2(f0.x, f0.y);
            afrag0[kc].pk[1] = pack2(f0.z, f0.w);
            afrag0[kc].pk[2] = pack2(f1.x, f1.y);
            afrag0[kc].pk[3] = pack2(f1.z, f1.w);
            afrag0[kc].pk[4] = pack2(f2.x, f2.y);
            afrag0[kc].pk[5] = pack2(f2.z, f2.w);
            afrag0[kc].pk[6] = pack2(f3.x, f3.y);
            afrag0[kc].pk[7] = pack2(f3.z, f3.w);
            f0 = vp1[c0];
            f1 = vp1[c0 + 1];
            f2 = vp1[c0 + 2];
            f3 = vp1[c0 + 3];
            afrag1[kc].pk[0] = pack2(f0.x, f0.y);
            afrag1[kc].pk[1] = pack2(f0.z, f0.w);
            afrag1[kc].pk[2] = pack2(f1.x, f1.y);
            afrag1[kc].pk[3] = pack2(f1.z, f1.w);
            afrag1[kc].pk[4] = pack2(f2.x, f2.y);
            afrag1[kc].pk[5] = pack2(f2.z, f2.w);
            afrag1[kc].pk[6] = pack2(f3.x, f3.y);
            afrag1[kc].pk[7] = pack2(f3.z, f3.w);
        }
    }

    // ---- loop over h-chunks: each B frag feeds two WMMAs, pipelined 1 deep --
    const v16bf* w1v = (const v16bf*)w1s;
    float p0[8], p1[8];
    #pragma unroll
    for (int r = 0; r < 8; ++r) { p0[r] = 0.f; p1[r] = 0.f; }

    v16bf cur = w1v[lh * 16 + grp];                               // hc=0, kc=0

    for (int hc = 0; hc < 16; ++hc) {
        const int base  = (hc * 16 + lh) * 16 + grp;              // kc=0 frag index
        const int nhc   = (hc < 15) ? hc + 1 : 15;                // clamped next hc
        const int nbase = (nhc * 16 + lh) * 16 + grp;

        v8f acc0 = {0.f, 0.f, 0.f, 0.f, 0.f, 0.f, 0.f, 0.f};
        v8f acc1 = {0.f, 0.f, 0.f, 0.f, 0.f, 0.f, 0.f, 0.f};
        #pragma unroll
        for (int kc = 0; kc < 8; ++kc) {
            const int ni = (kc < 7) ? (base + (kc + 1) * 2) : nbase;
            v16bf nxt = w1v[ni];                                  // issue next load first
            acc0 = __builtin_amdgcn_wmma_f32_16x16x32_bf16(
                false, afrag0[kc].v, false, cur, (short)0, acc0, false, false);
            acc1 = __builtin_amdgcn_wmma_f32_16x16x32_bf16(
                false, afrag1[kc].v, false, cur, (short)0, acc1, false, false);
            cur = nxt;
        }

        const int h  = hc * 16 + lh;
        const float qv = qs[h];
        const float wv = wsd[h];
        #pragma unroll
        for (int r = 0; r < 8; ++r) {
            p0[r] += TANHF(acc0[r] + qv) * wv;
            p1[r] += TANHF(acc1[r] + qv) * wv;
        }
    }

    // Reduce over N within each 16-lane half (xor masks 1,2,4,8 stay in-half).
    // C layout: VGPR r -> row r (lanes 0-15) / row 8+r (lanes 16-31).
    #pragma unroll
    for (int m = 1; m < 16; m <<= 1) {
        #pragma unroll
        for (int r = 0; r < 8; ++r) {
            p0[r] += __shfl_xor(p0[r], m, 32);
            p1[r] += __shfl_xor(p1[r], m, 32);
        }
    }
    if (lh == 0) {
        float* out0 = scores + (size_t)b * S_ + s0 + grp * 8;
        float* out1 = out0 + 16;
        #pragma unroll
        for (int r = 0; r < 8; ++r) { out0[r] = p0[r]; out1[r] = p1[r]; }
    }
}

// ---------------------------------------------------------------------------
// Kernel 3: per-batch masked softmax stats (max, sum-exp)
// ---------------------------------------------------------------------------
__global__ void __launch_bounds__(256) stats_kernel(const float* __restrict__ scores,
                                                    const int* __restrict__ lens,
                                                    float* __restrict__ stats) {
    __shared__ float red[256];
    const int b   = blockIdx.x;
    const int tid = threadIdx.x;
    const int len = lens[b];
    const float* sc = scores + (size_t)b * S_;

    float m = -INFINITY;
    for (int s = tid; s < len; s += 256) m = fmaxf(m, sc[s]);
    red[tid] = m;
    __syncthreads();
    for (int off = 128; off > 0; off >>= 1) {
        if (tid < off) red[tid] = fmaxf(red[tid], red[tid + off]);
        __syncthreads();
    }
    const float gm = red[0];
    __syncthreads();

    float d = 0.f;
    for (int s = tid; s < len; s += 256) d += __expf(sc[s] - gm);
    red[tid] = d;
    __syncthreads();
    for (int off = 128; off > 0; off >>= 1) {
        if (tid < off) red[tid] += red[tid + off];
        __syncthreads();
    }
    if (tid == 0) { stats[b * 2] = gm; stats[b * 2 + 1] = red[0]; }
}

// ---------------------------------------------------------------------------
// Kernel 4: partial weighted sums over 16 S-chunks (bandwidth-bound V re-read)
// ---------------------------------------------------------------------------
__global__ void __launch_bounds__(256) wsum_kernel(const float* __restrict__ V,
                                                   const float* __restrict__ scores,
                                                   const float* __restrict__ stats,
                                                   const int* __restrict__ lens,
                                                   float* __restrict__ part) {
    const int b  = blockIdx.y;
    const int s0 = blockIdx.x * 256;
    const int dv = threadIdx.x;
    const int len = lens[b];
    const float gm  = stats[b * 2];
    const float inv = 1.f / stats[b * 2 + 1];
    const float* Vb = V + (size_t)b * S_ * DV_;
    const float* sc = scores + (size_t)b * S_;

    float acc = 0.f;
    const int send = min(s0 + 256, len);
    for (int s = s0; s < send; ++s)
        acc += __expf(sc[s] - gm) * Vb[(size_t)s * DV_ + dv];
    part[((size_t)b * 16 + blockIdx.x) * DV_ + dv] = acc * inv;
}

// ---------------------------------------------------------------------------
// Kernel 5: deterministic fixed-order reduction of chunk partials
// ---------------------------------------------------------------------------
__global__ void __launch_bounds__(256) reduce_kernel(const float* __restrict__ part,
                                                     float* __restrict__ out) {
    const int b  = blockIdx.x;
    const int dv = threadIdx.x;
    float acc = 0.f;
    #pragma unroll
    for (int c = 0; c < 16; ++c)
        acc += part[((size_t)b * 16 + c) * DV_ + dv];
    out[b * DV_ + dv] = acc;
}

// ---------------------------------------------------------------------------
extern "C" void kernel_launch(void* const* d_in, const int* in_sizes, int n_in,
                              void* d_out, int out_size, void* d_ws, size_t ws_size,
                              hipStream_t stream) {
    const float* value = (const float*)d_in[0];   // [B,S,DV]
    const float* query = (const float*)d_in[1];   // [B,DQ]
    const int*   lens  = (const int*)d_in[2];     // [B]
    const float* W1    = (const float*)d_in[3];   // [H,DV]
    const float* W2    = (const float*)d_in[4];   // [H,DQ]
    const float* bias  = (const float*)d_in[5];   // [1,1,H]
    const float* wvec  = (const float*)d_in[6];   // [H]
    float* out = (float*)d_out;                   // [B,DV]

    char* ws = (char*)d_ws;
    float*          qpb    = (float*)(ws);                          //  32 KB
    unsigned short* W1bf   = (unsigned short*)(ws + 32768);         // 128 KB
    float*          scores = (float*)(ws + 32768 + 131072);         // 512 KB
    float*          stats  = (float*)(ws + 32768 + 131072 + 524288);//  256 B
    float*          part   = (float*)(ws + 32768 + 131072 + 524288 + 256); // 512 KB

    const size_t score_lds = (size_t)H_ * DV_ * 2 + 2 * H_ * sizeof(float); // 130 KB

    qpb_kernel<<<B_, 256, 0, stream>>>(query, W2, bias, qpb);
    convw1_kernel<<<(H_ * DV_) / (256 * 4), 256, 0, stream>>>(W1, W1bf);
    score_kernel<<<dim3(S_ / 512, B_), 512, score_lds, stream>>>(value, W1bf, qpb, wvec, scores);
    stats_kernel<<<B_, 256, 0, stream>>>(scores, lens, stats);
    wsum_kernel<<<dim3(S_ / 256, B_), 256, 0, stream>>>(value, scores, stats, lens, part);
    reduce_kernel<<<B_, 256, 0, stream>>>(part, out);
}
